// Net_14937896256212
// MI455X (gfx1250) — matrix operand order repaired and checked by reading, not compiled, not run on hardware
//
#include <hip/hip_runtime.h>
#include <hip/hip_bf16.h>

// ---------------------------------------------------------------------------
// SAGPool GNN forward for MI455X (gfx1250, wave32).
// GEMMs: v_wmma_f32_16x16x32_bf16, 64-row blocks, 4 accumulators/wave,
// W pre-packed into fragment order (contiguous 32B per-lane B fragments).
// Scatter/gather phases: float4 gathers + global_atomic_add_f32 streams.
// ---------------------------------------------------------------------------

typedef __attribute__((ext_vector_type(16))) __bf16 v16bf;
typedef __attribute__((ext_vector_type(4)))  __bf16 v4bf;
typedef __attribute__((ext_vector_type(8)))  float  v8f;

static constexpr int       Gn   = 512;
static constexpr int       NPG  = 200;
static constexpr int       K1   = 100;
static constexpr int       K2   = 50;
static constexpr long long Nn   = (long long)Gn * NPG;   // 102400 nodes
static constexpr long long Ee   = Nn * 12;               // 1228800 edges
static constexpr long long N1   = (long long)Gn * K1;    // 51200
static constexpr long long N2   = (long long)Gn * K2;    // 25600
static constexpr int       DIM  = 64;
static constexpr int       INCH = 128;
static constexpr int       NCLS = 10;
static constexpr float     BN_EPS = 1e-5f;

static inline unsigned blocks(long long total, int bs) {
  return (unsigned)((total + bs - 1) / bs);
}

// ------------------------------- elementwise -------------------------------

__global__ void k_fill(float* p, float v, long long n) {
  long long i = blockIdx.x * (long long)blockDim.x + threadIdx.x;
  if (i < n) p[i] = v;
}

// Pack W[K,64] fp32 -> bf16 WMMA-B fragment order:
// P[((tile*(K/32)+kb)*32 + lane)*16 + i] = W[(kb*32 + (lane>>4)*16 + i)*64 + tile*16 + (lane&15)]
__global__ void k_pack_w(const float* __restrict__ W, __bf16* __restrict__ P, int K) {
  int idx = blockIdx.x * blockDim.x + threadIdx.x;
  if (idx >= K * 64) return;
  int i    = idx & 15;
  int lane = (idx >> 4) & 31;
  int t    = idx >> 9;            // tile*(K/32) + kb
  int kb   = t % (K / 32);
  int tile = t / (K / 32);
  int k    = kb * 32 + (lane >> 4) * 16 + i;
  int col  = tile * 16 + (lane & 15);
  P[idx] = (__bf16)W[k * 64 + col];
}

// ------------------------- edge gather / scatter-add -----------------------
// float4 gather (global_load_b128) + 4 float atomics per thread.
__global__ void k_scatter_feat(const float* __restrict__ x, const int* __restrict__ src,
                               const int* __restrict__ dst, const float* __restrict__ m,
                               float* __restrict__ out, long long E, int dimv /*dim/4*/) {
  long long tid = blockIdx.x * (long long)blockDim.x + threadIdx.x;
  if (tid >= E * dimv) return;
  long long e = tid / dimv;
  int c = (int)(tid % dimv);
  float w = m[e];
  if (w == 0.f) return;
  const float4 v = ((const float4*)x)[(size_t)src[e] * dimv + c];
  float* o = out + ((size_t)dst[e] * dimv + c) * 4;
  atomicAdd(o + 0, v.x * w);
  atomicAdd(o + 1, v.y * w);
  atomicAdd(o + 2, v.z * w);
  atomicAdd(o + 3, v.w * w);
}

__global__ void k_scatter_1d(const int* __restrict__ dst, const float* __restrict__ m,
                             float* __restrict__ deg, long long E) {
  long long e = blockIdx.x * (long long)blockDim.x + threadIdx.x;
  if (e < E && m[e] != 0.f) atomicAdd(&deg[dst[e]], m[e]);
}

// ------------------------------ WMMA GEMM ----------------------------------
// C[row0..row0+63, 0..63] = act((A (+Aadd)) @ W + bias)
// 128 threads = 4 waves; wave w owns columns [16w,16w+16) and 4 row sub-tiles.
// A (fp32, optionally fused x+msg) staged to LDS as bf16 with float4 loads.
// Wp is the fragment-packed bf16 weight (one 32B contiguous chunk per lane).
template<int K>
__global__ void k_gemm_wmma(const float* __restrict__ A, const float* __restrict__ Aadd,
                            const __bf16* __restrict__ Wp, const float* __restrict__ bias,
                            float* __restrict__ C, int relu) {
  __shared__ __bf16 sA[64 * K];
  const int row0 = blockIdx.x * 64;

  for (int i4 = threadIdx.x; i4 < 64 * K / 4; i4 += 128) {
    int i = i4 * 4;
    int r = i / K, k = i % K;           // K % 4 == 0: chunk never crosses a row
    size_t gi = (size_t)(row0 + r) * K + k;
    float4 v = *(const float4*)&A[gi];
    if (Aadd) {
      float4 u = *(const float4*)&Aadd[gi];
      v.x += u.x; v.y += u.y; v.z += u.z; v.w += u.w;
    }
    *(v4bf*)&sA[i] = (v4bf){(__bf16)v.x, (__bf16)v.y, (__bf16)v.z, (__bf16)v.w};
  }
  __syncthreads();

  const int wave = threadIdx.x >> 5;
  const int lane = threadIdx.x & 31;
  const int l    = lane & 15;
  const int hi   = lane >> 4;

  v8f acc[4] = {};
  #pragma unroll
  for (int kb32 = 0; kb32 < K / 32; ++kb32) {
    // B fragment: one contiguous 32-byte chunk per lane.
    const v16bf b = *(const v16bf*)(Wp + ((size_t)(wave * (K / 32) + kb32) * 32 + lane) * 16);
    #pragma unroll
    for (int rt = 0; rt < 4; ++rt) {
      v16bf a;
      #pragma unroll
      for (int i = 0; i < 16; ++i) {
        // A 16x32 bf16 layout: K=hi*16+i (contiguous, matches packed B)
        int ka = kb32 * 32 + hi * 16 + i;
        a[i] = sA[(rt * 16 + l) * K + ka];
      }
      acc[rt] = __builtin_amdgcn_wmma_f32_16x16x32_bf16(
          false, a, false, b, (short)0, acc[rt], false, false);
    }
  }

  const int n = wave * 16 + l;
  const float bn = bias[n];
  #pragma unroll
  for (int rt = 0; rt < 4; ++rt) {
    #pragma unroll
    for (int r = 0; r < 8; ++r) {
      int mrow = rt * 16 + (hi ? (r + 8) : r);   // C/D: VGPR r -> M=r / M=r+8
      float v = acc[rt][r] + bn;
      if (relu) v = fmaxf(v, 0.f);
      C[(size_t)(row0 + mrow) * DIM + n] = v;
    }
  }
}

// ----------------------------- GCN score (dim 1) ---------------------------

__global__ void k_dot64(const float* __restrict__ x, const float* __restrict__ w,
                        float* __restrict__ xw, long long n) {
  long long i = blockIdx.x * (long long)blockDim.x + threadIdx.x;
  if (i >= n) return;
  float s = 0.f;
  #pragma unroll
  for (int k = 0; k < DIM; ++k) s += x[i * DIM + k] * w[k];
  xw[i] = s;
}

__global__ void k_invsqrt_deg(float* deg, long long n) {
  long long i = blockIdx.x * (long long)blockDim.x + threadIdx.x;
  if (i < n) deg[i] = rsqrtf(deg[i] + 1.f);   // deg -> deg^{-1/2} with self-loop
}

__global__ void k_scatter_agg(const int* __restrict__ src, const int* __restrict__ dst,
                              const float* __restrict__ m, const float* __restrict__ xw,
                              const float* __restrict__ dis, float* __restrict__ agg,
                              long long E) {
  long long e = blockIdx.x * (long long)blockDim.x + threadIdx.x;
  if (e >= E) return;
  float w = m[e];
  if (w != 0.f) atomicAdd(&agg[dst[e]], xw[src[e]] * dis[src[e]] * w);
}

__global__ void k_score(const float* __restrict__ dis, const float* __restrict__ agg,
                        const float* __restrict__ xw, const float* __restrict__ bg,
                        float* __restrict__ sc, long long n) {
  long long i = blockIdx.x * (long long)blockDim.x + threadIdx.x;
  if (i < n) sc[i] = dis[i] * agg[i] + dis[i] * dis[i] * xw[i] + bg[0];
}

// ------------------------------- top-k pooling -----------------------------

__global__ void k_topk(const float* __restrict__ score, int npg, int k,
                       int* __restrict__ perm) {
  __shared__ float s[256];
  int g = blockIdx.x, t = threadIdx.x;
  if (t < npg) s[t] = score[(size_t)g * npg + t];
  __syncthreads();
  if (t < npg) {
    float mi = s[t];
    int rank = 0;
    for (int j = 0; j < npg; ++j) {
      float mj = s[j];
      if (mj > mi || (mj == mi && j < t)) rank++;
    }
    if (rank < k) perm[g * k + rank] = g * npg + t;   // descending, ties by index
  }
}

__global__ void k_pool_feat(const float* __restrict__ x, const float* __restrict__ score,
                            const int* __restrict__ perm, float* __restrict__ xn,
                            long long total) {
  long long tid = blockIdx.x * (long long)blockDim.x + threadIdx.x;
  if (tid >= total) return;
  long long node = tid / DIM;
  int f = (int)(tid % DIM);
  int p = perm[node];
  xn[tid] = x[(size_t)p * DIM + f] * tanhf(score[p]);
}

__global__ void k_set_inv(const int* __restrict__ perm, int* __restrict__ inv, long long nk) {
  long long j = blockIdx.x * (long long)blockDim.x + threadIdx.x;
  if (j < nk) inv[perm[j]] = (int)j;
}

__global__ void k_relabel(const int* __restrict__ src, const int* __restrict__ dst,
                          const float* __restrict__ m, const int* __restrict__ inv,
                          int* __restrict__ ns, int* __restrict__ nd,
                          float* __restrict__ nm, long long E) {
  long long e = blockIdx.x * (long long)blockDim.x + threadIdx.x;
  if (e >= E) return;
  int a = inv[src[e]], b = inv[dst[e]];
  nm[e] = m[e] * ((a >= 0 && b >= 0) ? 1.f : 0.f);
  ns[e] = a > 0 ? a : 0;
  nd[e] = b > 0 ? b : 0;
}

// --------------------------------- BatchNorm -------------------------------

__global__ void k_bn_stats(const float* __restrict__ x, int n, int d,
                           float* __restrict__ mean, float* __restrict__ var) {
  __shared__ float ssum[256], ssq[256];
  int c = blockIdx.x;
  float s = 0.f, q = 0.f;
  for (int i = threadIdx.x; i < n; i += blockDim.x) {
    float v = x[(size_t)i * d + c];
    s += v; q += v * v;
  }
  ssum[threadIdx.x] = s; ssq[threadIdx.x] = q;
  __syncthreads();
  for (int st = 128; st > 0; st >>= 1) {
    if ((int)threadIdx.x < st) {
      ssum[threadIdx.x] += ssum[threadIdx.x + st];
      ssq[threadIdx.x]  += ssq[threadIdx.x + st];
    }
    __syncthreads();
  }
  if (threadIdx.x == 0) {
    float mu = ssum[0] / n;
    mean[c] = mu;
    var[c]  = ssq[0] / n - mu * mu;
  }
}

__global__ void k_bn_apply(const float* __restrict__ x, const float* __restrict__ mean,
                           const float* __restrict__ var, const float* __restrict__ g,
                           const float* __restrict__ b, float* __restrict__ y,
                           long long total, int d) {
  long long tid = blockIdx.x * (long long)blockDim.x + threadIdx.x;
  if (tid >= total) return;
  int c = (int)(tid % d);
  y[tid] = (x[tid] - mean[c]) * rsqrtf(var[c] + BN_EPS) * g[c] + b[c];
}

// ---------------------------------- unpool ---------------------------------

__global__ void k_set_sel(const int* __restrict__ perm, float* __restrict__ sel, long long nk) {
  long long j = blockIdx.x * (long long)blockDim.x + threadIdx.x;
  if (j < nk) sel[perm[j]] = 1.f;
}

__global__ void k_set_up(const float* __restrict__ xs, const int* __restrict__ perm,
                         float* __restrict__ up, long long total) {
  long long tid = blockIdx.x * (long long)blockDim.x + threadIdx.x;
  if (tid >= total) return;
  long long node = tid / DIM;
  int f = (int)(tid % DIM);
  up[(size_t)perm[node] * DIM + f] = xs[tid];
}

__global__ void k_unpool_feat(const int* __restrict__ src, const int* __restrict__ dst,
                              const float* __restrict__ m, const float* __restrict__ sel,
                              const float* __restrict__ up, float* __restrict__ ssum,
                              long long E) {
  long long tid = blockIdx.x * (long long)blockDim.x + threadIdx.x;
  if (tid >= E * (DIM / 4)) return;
  long long e = tid / (DIM / 4);
  int c = (int)(tid % (DIM / 4));
  float w = m[e] * sel[src[e]];
  if (w == 0.f) return;
  const float4 v = ((const float4*)up)[(size_t)src[e] * (DIM / 4) + c];
  float* o = ssum + ((size_t)dst[e] * (DIM / 4) + c) * 4;
  atomicAdd(o + 0, v.x * w);
  atomicAdd(o + 1, v.y * w);
  atomicAdd(o + 2, v.z * w);
  atomicAdd(o + 3, v.w * w);
}

__global__ void k_unpool_cnt(const int* __restrict__ src, const int* __restrict__ dst,
                             const float* __restrict__ m, const float* __restrict__ sel,
                             float* __restrict__ cnt, long long E) {
  long long e = blockIdx.x * (long long)blockDim.x + threadIdx.x;
  if (e >= E) return;
  float w = m[e] * sel[src[e]];
  if (w != 0.f) atomicAdd(&cnt[dst[e]], w);
}

__global__ void k_unpool_final(const float* __restrict__ sel, const float* __restrict__ up,
                               const float* __restrict__ ssum, const float* __restrict__ cnt,
                               const float* __restrict__ rw, const float* __restrict__ add,
                               float* __restrict__ out, long long total) {
  long long tid = blockIdx.x * (long long)blockDim.x + threadIdx.x;
  if (tid >= total) return;
  long long i = tid / DIM;
  float base = (sel[i] > 0.f) ? up[tid] : ssum[tid] / fmaxf(cnt[i], 1.f);
  out[tid] = base + rw[0] * add[tid];
}

// --------------------------------- readout ---------------------------------

__global__ void k_readout(const float* __restrict__ x, int k, float* __restrict__ r, int off) {
  int g = blockIdx.x, d = threadIdx.x;   // blockDim == DIM
  float mx = -INFINITY, sm = 0.f;
  for (int i = 0; i < k; ++i) {
    float v = x[((size_t)g * k + i) * DIM + d];
    mx = fmaxf(mx, v); sm += v;
  }
  r[(size_t)g * 384 + off + d]       = mx;
  r[(size_t)g * 384 + off + DIM + d] = sm / k;
}

__global__ void k_attn(const float* __restrict__ r, const float* __restrict__ Wa,
                       const float* __restrict__ ba, float* __restrict__ rr, int gcount) {
  int g = blockIdx.x * blockDim.x + threadIdx.x;
  if (g >= gcount) return;
  float lg[3];
  for (int j = 0; j < 3; ++j) {
    float s = ba[j];
    for (int k = 0; k < 384; ++k) s += r[(size_t)g * 384 + k] * Wa[k * 3 + j];
    lg[j] = s;
  }
  float mx = fmaxf(lg[0], fmaxf(lg[1], lg[2]));
  float e0 = expf(lg[0] - mx), e1 = expf(lg[1] - mx), e2 = expf(lg[2] - mx);
  float inv = 1.f / (e0 + e1 + e2);
  float a[3] = {e0 * inv, e1 * inv, e2 * inv};
  for (int k = 0; k < 384; ++k) rr[(size_t)g * 384 + k] = r[(size_t)g * 384 + k] * a[k / 128];
}

__global__ void k_final(const float* __restrict__ r, const float* __restrict__ Wl,
                        const float* __restrict__ bl, float* __restrict__ out, int gcount) {
  int g = blockIdx.x * blockDim.x + threadIdx.x;
  if (g >= gcount) return;
  float o[NCLS];
  float mx = -INFINITY;
  for (int j = 0; j < NCLS; ++j) {
    float s = bl[j];
    for (int k = 0; k < 384; ++k) s += r[(size_t)g * 384 + k] * Wl[k * NCLS + j];
    o[j] = fmaxf(s, 0.f);
    mx = fmaxf(mx, o[j]);
  }
  float lse = 0.f;
  for (int j = 0; j < NCLS; ++j) lse += expf(o[j] - mx);
  lse = logf(lse) + mx;
  for (int j = 0; j < NCLS; ++j) out[(size_t)g * NCLS + j] = o[j] - lse;
}

// ================================ launcher =================================

extern "C" void kernel_launch(void* const* d_in, const int* in_sizes, int n_in,
                              void* d_out, int out_size, void* d_ws, size_t ws_size,
                              hipStream_t stream) {
  (void)in_sizes; (void)n_in; (void)out_size; (void)ws_size;

  // JAX pytree flatten: top-level dict insertion order; params dict key-sorted:
  // attn, bn1..bn6, c1..c5, g1, g2, lin1, rw1, rw2  ->  45 flat inputs.
  const float* X    = (const float*)d_in[0];
  const int*   SRC  = (const int*)  d_in[1];
  const int*   DST  = (const int*)  d_in[2];
  const float* AT_W = (const float*)d_in[3];
  const float* AT_B = (const float*)d_in[4];
  const float *BNg[6], *BNb[6];
  for (int i = 0; i < 6; ++i) { BNg[i] = (const float*)d_in[5 + 2 * i];
                                BNb[i] = (const float*)d_in[6 + 2 * i]; }
  const float *CW1[5], *CB1[5], *CW2[5], *CB2[5];
  for (int i = 0; i < 5; ++i) {
    CW1[i] = (const float*)d_in[17 + 4 * i]; CB1[i] = (const float*)d_in[18 + 4 * i];
    CW2[i] = (const float*)d_in[19 + 4 * i]; CB2[i] = (const float*)d_in[20 + 4 * i];
  }
  const float* G1W = (const float*)d_in[37]; const float* G1B = (const float*)d_in[38];
  const float* G2W = (const float*)d_in[39]; const float* G2B = (const float*)d_in[40];
  const float* LW  = (const float*)d_in[41]; const float* LB  = (const float*)d_in[42];
  const float* RW1 = (const float*)d_in[43]; const float* RW2 = (const float*)d_in[44];

  // ------------------------- workspace bump allocator ----------------------
  char* wsp = (char*)d_ws; size_t off = 0;
  auto alloc = [&](size_t bytes) -> char* {
    char* p = wsp + off; off += (bytes + 255) & ~(size_t)255; return p;
  };
  float* m0   = (float*)alloc(Ee * 4);
  float* msg  = (float*)alloc(Nn * INCH * 4);   // GIN scatter target (max dim)
  float* hid  = (float*)alloc(Nn * DIM * 4);    // GIN hidden
  float* x0   = (float*)alloc(Nn * DIM * 4);
  float* score= (float*)alloc(Nn * 4);
  float* xw   = (float*)alloc(Nn * 4);
  float* deg  = (float*)alloc(Nn * 4);
  float* agg  = (float*)alloc(Nn * 4);
  int*   perm1= (int*)alloc(N1 * 4);
  int*   perm2= (int*)alloc(N2 * 4);
  int*   inv  = (int*)alloc(Nn * 4);
  int*   s1 = (int*)alloc(Ee * 4); int* d1 = (int*)alloc(Ee * 4); float* m1 = (float*)alloc(Ee * 4);
  int*   s2 = (int*)alloc(Ee * 4); int* d2 = (int*)alloc(Ee * 4); float* m2 = (float*)alloc(Ee * 4);
  float* x1p = (float*)alloc(N1 * DIM * 4);
  float* x1  = (float*)alloc(N1 * DIM * 4);
  float* xg1 = (float*)alloc(N1 * DIM * 4);
  float* x2p = (float*)alloc(N2 * DIM * 4);
  float* x2  = (float*)alloc(N2 * DIM * 4);
  float* x2g = (float*)alloc(N2 * DIM * 4);
  float* x2b = (float*)alloc(N2 * DIM * 4);
  float* x3u = (float*)alloc(N1 * DIM * 4);
  float* x3g = (float*)alloc(N1 * DIM * 4);
  float* x3b = (float*)alloc(N1 * DIM * 4);
  float* x4u = (float*)alloc(Nn * DIM * 4);
  float* x4g = (float*)alloc(Nn * DIM * 4);
  float* x4b = (float*)alloc(Nn * DIM * 4);
  float* sel = (float*)alloc(Nn * 4);
  float* cnt = (float*)alloc(Nn * 4);
  float* up   = msg;                              // alias: unpool temps reuse GIN temp
  float* ssum = (float*)alloc(Nn * DIM * 4);
  float* r  = (float*)alloc((size_t)Gn * 384 * 4);
  float* rr = (float*)alloc((size_t)Gn * 384 * 4);
  float* rb = (float*)alloc((size_t)Gn * 384 * 4);
  float* stm = (float*)alloc(384 * 4);
  float* stv = (float*)alloc(384 * 4);
  __bf16* Wb[10];
  Wb[0] = (__bf16*)alloc((size_t)INCH * DIM * 2);
  for (int i = 1; i < 10; ++i) Wb[i] = (__bf16*)alloc((size_t)DIM * DIM * 2);

  // ------------------------- weight fragment packing -----------------------
  k_pack_w<<<blocks(INCH * DIM, 256), 256, 0, stream>>>(CW1[0], Wb[0], INCH);
  k_pack_w<<<blocks(DIM * DIM, 256), 256, 0, stream>>>(CW2[0], Wb[1], DIM);
  for (int i = 1; i < 5; ++i) {
    k_pack_w<<<blocks(DIM * DIM, 256), 256, 0, stream>>>(CW1[i], Wb[2 * i],     DIM);
    k_pack_w<<<blocks(DIM * DIM, 256), 256, 0, stream>>>(CW2[i], Wb[2 * i + 1], DIM);
  }

  // ------------------------------ helper passes ----------------------------
  auto run_gin = [&](const float* xin, const int* es, const int* ed, const float* em,
                     long long n, int di, const __bf16* W1p, const float* b1,
                     const __bf16* W2p, const float* b2, float* out) {
    hipMemsetAsync(msg, 0, (size_t)n * di * 4, stream);
    k_scatter_feat<<<blocks(Ee * (di / 4), 256), 256, 0, stream>>>(xin, es, ed, em, msg,
                                                                   Ee, di / 4);
    // h = x + msg fused into GEMM1's LDS staging
    if (di == INCH)
      k_gemm_wmma<INCH><<<(unsigned)(n / 64), 128, 0, stream>>>(xin, msg, W1p, b1, hid, 1);
    else
      k_gemm_wmma<DIM><<<(unsigned)(n / 64), 128, 0, stream>>>(xin, msg, W1p, b1, hid, 1);
    // outer relu of forward() fused into the second linear (all 5 convs are relu-wrapped)
    k_gemm_wmma<DIM><<<(unsigned)(n / 64), 128, 0, stream>>>(hid, nullptr, W2p, b2, out, 1);
  };

  auto run_score = [&](const float* xin, const int* es, const int* ed, const float* em,
                       long long n, const float* Wg, const float* bg, float* sc) {
    k_dot64<<<blocks(n, 256), 256, 0, stream>>>(xin, Wg, xw, n);
    hipMemsetAsync(deg, 0, (size_t)n * 4, stream);
    k_scatter_1d<<<blocks(Ee, 256), 256, 0, stream>>>(ed, em, deg, Ee);
    k_invsqrt_deg<<<blocks(n, 256), 256, 0, stream>>>(deg, n);
    hipMemsetAsync(agg, 0, (size_t)n * 4, stream);
    k_scatter_agg<<<blocks(Ee, 256), 256, 0, stream>>>(es, ed, em, xw, deg, agg, Ee);
    k_score<<<blocks(n, 256), 256, 0, stream>>>(deg, agg, xw, bg, sc, n);
  };

  auto run_bn = [&](const float* xin, long long n, int d, const float* g,
                    const float* b, float* y) {
    k_bn_stats<<<d, 256, 0, stream>>>(xin, (int)n, d, stm, stv);
    k_bn_apply<<<blocks(n * d, 256), 256, 0, stream>>>(xin, stm, stv, g, b, y, n * d, d);
  };

  auto run_unpool = [&](const float* xs, const int* es, const int* ed, const float* em,
                        const int* pm, long long nk, long long nbig,
                        const float* rw, const float* addend, float* out) {
    hipMemsetAsync(sel, 0, (size_t)nbig * 4, stream);
    k_set_sel<<<blocks(nk, 256), 256, 0, stream>>>(pm, sel, nk);
    hipMemsetAsync(up, 0, (size_t)nbig * DIM * 4, stream);
    k_set_up<<<blocks(nk * DIM, 256), 256, 0, stream>>>(xs, pm, up, nk * DIM);
    hipMemsetAsync(ssum, 0, (size_t)nbig * DIM * 4, stream);
    hipMemsetAsync(cnt, 0, (size_t)nbig * 4, stream);
    k_unpool_feat<<<blocks(Ee * (DIM / 4), 256), 256, 0, stream>>>(es, ed, em, sel, up,
                                                                   ssum, Ee);
    k_unpool_cnt<<<blocks(Ee, 256), 256, 0, stream>>>(es, ed, em, sel, cnt, Ee);
    k_unpool_final<<<blocks(nbig * DIM, 256), 256, 0, stream>>>(sel, up, ssum, cnt, rw,
                                                                addend, out, nbig * DIM);
  };

  // --------------------------------- forward -------------------------------
  k_fill<<<blocks(Ee, 256), 256, 0, stream>>>(m0, 1.f, Ee);

  // x0 = relu(gin_c1(x))
  run_gin(X, SRC, DST, m0, Nn, INCH, Wb[0], CB1[0], Wb[1], CB2[0], x0);

  // sag_pool 1: (x0, src, dst, m0) -> x1p, s1, d1, m1, perm1
  run_score(x0, SRC, DST, m0, Nn, G1W, G1B, score);
  k_topk<<<Gn, 256, 0, stream>>>(score, NPG, K1, perm1);
  k_pool_feat<<<blocks(N1 * DIM, 256), 256, 0, stream>>>(x0, score, perm1, x1p, N1 * DIM);
  hipMemsetAsync(inv, 0xFF, (size_t)Nn * 4, stream);
  k_set_inv<<<blocks(N1, 256), 256, 0, stream>>>(perm1, inv, N1);
  k_relabel<<<blocks(Ee, 256), 256, 0, stream>>>(SRC, DST, m0, inv, s1, d1, m1, Ee);

  run_bn(x1p, N1, DIM, BNg[0], BNb[0], x1);                            // x1 = bn1(x1p)
  run_gin(x1, s1, d1, m1, N1, DIM, Wb[2], CB1[1], Wb[3], CB2[1], xg1); // relu(gin_c2)

  // sag_pool 2: (xg1, s1, d1, m1) -> x2p, s2, d2, m2, perm2
  run_score(xg1, s1, d1, m1, N1, G2W, G2B, score);
  k_topk<<<Gn, 256, 0, stream>>>(score, K1, K2, perm2);
  k_pool_feat<<<blocks(N2 * DIM, 256), 256, 0, stream>>>(xg1, score, perm2, x2p, N2 * DIM);
  hipMemsetAsync(inv, 0xFF, (size_t)N1 * 4, stream);
  k_set_inv<<<blocks(N2, 256), 256, 0, stream>>>(perm2, inv, N2);
  k_relabel<<<blocks(Ee, 256), 256, 0, stream>>>(s1, d1, m1, inv, s2, d2, m2, Ee);

  run_bn(x2p, N2, DIM, BNg[1], BNb[1], x2);                            // bn2
  run_gin(x2, s2, d2, m2, N2, DIM, Wb[4], CB1[2], Wb[5], CB2[2], x2g); // relu(gin_c3)
  run_bn(x2g, N2, DIM, BNg[2], BNb[2], x2b);                           // bn3

  // x3 = unpool(x2b -> N1) + rw1 * xg1 ; relu(gin_c4) ; bn4
  run_unpool(x2b, s1, d1, m1, perm2, N2, N1, RW1, xg1, x3u);
  run_gin(x3u, s1, d1, m1, N1, DIM, Wb[6], CB1[3], Wb[7], CB2[3], x3g);
  run_bn(x3g, N1, DIM, BNg[3], BNb[3], x3b);

  // x4 = unpool(x3b -> N) + rw2 * x0 ; relu(gin_c5) ; bn5
  run_unpool(x3b, SRC, DST, m0, perm1, N1, Nn, RW2, x0, x4u);
  run_gin(x4u, SRC, DST, m0, Nn, DIM, Wb[8], CB1[4], Wb[9], CB2[4], x4g);
  run_bn(x4g, Nn, DIM, BNg[4], BNb[4], x4b);

  // readout -> r[G,384] = [gmp||gap](x2b) || [..](x3b) || [..](x4b)
  k_readout<<<Gn, DIM, 0, stream>>>(x2b, K2, r, 0);
  k_readout<<<Gn, DIM, 0, stream>>>(x3b, K1, r, 128);
  k_readout<<<Gn, DIM, 0, stream>>>(x4b, NPG, r, 256);

  // attention gating, bn6, final linear + relu + log_softmax
  k_attn<<<blocks(Gn, 128), 128, 0, stream>>>(r, AT_W, AT_B, rr, Gn);
  run_bn(rr, Gn, 384, BNg[5], BNb[5], rb);
  k_final<<<blocks(Gn, 128), 128, 0, stream>>>(rb, LW, LB, (float*)d_out, Gn);
}
